// MixEncoderLayer_10273561772125
// MI455X (gfx1250) — matrix operand order repaired and checked by reading, not compiled, not run on hardware
//
#include <hip/hip_runtime.h>
#include <hip/hip_bf16.h>
#include <stdint.h>

// ---------------------------------------------------------------------------
// Types
// ---------------------------------------------------------------------------
typedef __bf16 bf16_t;
typedef __attribute__((ext_vector_type(16))) __bf16 v16bf;
typedef __attribute__((ext_vector_type(8)))  float  v8f;
typedef __attribute__((ext_vector_type(4)))  unsigned int u32x4;

static __device__ __forceinline__ bf16_t f2bf(float f) {
    // round-to-nearest-even f32 -> bf16
    uint32_t u = __builtin_bit_cast(uint32_t, f);
    uint32_t r = u + 0x7FFFu + ((u >> 16) & 1u);
    uint16_t h = (uint16_t)(r >> 16);
    return __builtin_bit_cast(bf16_t, h);
}

static __device__ __forceinline__ v8f wmma_bf16(v16bf a, v16bf b, v8f c) {
    return __builtin_amdgcn_wmma_f32_16x16x32_bf16(
        /*neg_a=*/false, a, /*neg_b=*/false, b,
        /*c_mod=*/(short)0, c, /*reuse_a=*/false, /*reuse_b=*/false);
}

// A-fragment (16x32 MxK, bf16, row-major source, leading dim lda elements).
// Per ISA: lane L holds row M=L&15; elements 0..7 are K = off+0..7,
// elements 8..15 are K = off+16..23, where off = 8 if lane>=16.
static __device__ __forceinline__ v16bf load_a(const bf16_t* A, int lda,
                                               int m0, int k0, int lane) {
    int m   = m0 + (lane & 15);
    int off = (lane & 16) ? 8 : 0;
    const bf16_t* p = A + (size_t)m * lda + k0 + off;
    union { v16bf v; u32x4 q[2]; } f;
    f.q[0] = *(const u32x4*)(p);
    f.q[1] = *(const u32x4*)(p + 16);
    return f.v;
}

// B-fragment (32x16 KxN) built from a row-major weight W[N,K]:
// B[k][n] = W[n][k]. Lane L holds column n = L&15; elements 0..7 are
// K = koff+0..7, elements 8..15 are K = koff+16..23, koff = 8 if lane>=16.
static __device__ __forceinline__ v16bf load_b(const bf16_t* W, int ldb,
                                               int n0, int k0, int lane) {
    int n    = n0 + (lane & 15);
    int koff = (lane & 16) ? 8 : 0;
    const bf16_t* p = W + (size_t)n * ldb + k0 + koff;
    union { v16bf v; u32x4 q[2]; } f;
    f.q[0] = *(const u32x4*)(p);
    f.q[1] = *(const u32x4*)(p + 16);
    return f.v;
}

// A-fragment from the in-place LDS probability stripe. Each 4096-byte f32 row
// holds bf16 probs: cols [0,512) at bf16 index [0,512), cols [512,1024) at
// bf16 index [1024,1536). Row stride = 2048 bf16 slots.
static __device__ __forceinline__ v16bf load_a_lds(const bf16_t* pb, int k0, int lane) {
    int ml  = lane & 15;
    int off = (lane & 16) ? 8 : 0;
    int c0  = k0 + off;
    int c1  = c0 + 16;
    int i0  = c0 + ((c0 >= 512) ? 512 : 0);
    int i1  = c1 + ((c1 >= 512) ? 512 : 0);
    const bf16_t* rowp = pb + ml * 2048;
    union { v16bf v; u32x4 q[2]; } f;
    f.q[0] = *(const u32x4*)(rowp + i0);
    f.q[1] = *(const u32x4*)(rowp + i1);
    return f.v;
}

// ---------------------------------------------------------------------------
// Elementwise helpers
// ---------------------------------------------------------------------------
__global__ void cast_f32_bf16(const float* __restrict__ in,
                              bf16_t* __restrict__ out, int n) {
    int i = blockIdx.x * 256 + threadIdx.x;
    if (i < n) out[i] = f2bf(in[i]);
}

// vb [B,S, H*64] bf16 -> vt [B,H,64,S] bf16 (coalesced writes)
__global__ void transpose_v(const bf16_t* __restrict__ vb,
                            bf16_t* __restrict__ vt) {
    int i  = blockIdx.x * 256 + threadIdx.x;   // 4,194,304 total
    int s  = i & 1023;
    int d  = (i >> 10) & 63;
    int bh = i >> 16;                          // b*8 + h
    int h  = bh & 7, b = bh >> 3;
    vt[i] = vb[((size_t)((b << 10) | s)) * 512 + ((h << 6) | d)];
}

// ---------------------------------------------------------------------------
// GEMM: C[M,N] = act( alpha * A[M,K] x W[N,K]^T + bias[n] + res[m,n] )
// block = 256 (8 waves); wave -> 16x64 strip; grid = (N/64, M/128)
// ---------------------------------------------------------------------------
__global__ __launch_bounds__(256) void gemm_bf16_wmma(
    const bf16_t* __restrict__ A, const bf16_t* __restrict__ W,
    const float* __restrict__ bias, const float* __restrict__ res,
    float* __restrict__ outf, bf16_t* __restrict__ outb,
    int M, int N, int K, int relu, float alpha)
{
    const int lane = threadIdx.x & 31;
    const int wave = threadIdx.x >> 5;
    const int m0   = blockIdx.y * 128 + wave * 16;
    const int n0   = blockIdx.x * 64;
    if (m0 >= M) return;

    v8f acc0 = {}, acc1 = {}, acc2 = {}, acc3 = {};
    for (int k0 = 0; k0 < K; k0 += 32) {
        v16bf a  = load_a(A, K, m0, k0, lane);
        v16bf b0 = load_b(W, K, n0 +  0, k0, lane);
        v16bf b1 = load_b(W, K, n0 + 16, k0, lane);
        v16bf b2 = load_b(W, K, n0 + 32, k0, lane);
        v16bf b3 = load_b(W, K, n0 + 48, k0, lane);
        acc0 = wmma_bf16(a, b0, acc0);
        acc1 = wmma_bf16(a, b1, acc1);
        acc2 = wmma_bf16(a, b2, acc2);
        acc3 = wmma_bf16(a, b3, acc3);
    }

    const int nc = n0 + (lane & 15);
    const int mb = m0 + ((lane >> 4) << 3);
    v8f accs[4] = {acc0, acc1, acc2, acc3};
#pragma unroll
    for (int t = 0; t < 4; ++t) {
        int n = nc + t * 16;
        float bv = bias ? bias[n] : 0.0f;
#pragma unroll
        for (int r = 0; r < 8; ++r) {
            int m = mb + r;
            float val = accs[t][r] * alpha + bv;
            if (res)  val += res[(size_t)m * N + n];
            if (relu) val = fmaxf(val, 0.0f);
            size_t o = (size_t)m * N + n;
            if (outf) outf[o] = val;
            if (outb) outb[o] = f2bf(val);
        }
    }
}

// ---------------------------------------------------------------------------
// Fused attention: one wave per (b, h, 16-query tile).
// scores(16x1024, f32, LDS) = (Q*0.125) K^T + cost ; softmax ; ctx = P V.
// Probabilities are converted bf16 in place (write ptr trails read ptr per
// lane; volatile DS ops + same-wave in-order DS make this safe).
// ---------------------------------------------------------------------------
__global__ __launch_bounds__(32) void attn_fused(
    const bf16_t* __restrict__ qb, const bf16_t* __restrict__ kb,
    const bf16_t* __restrict__ vt, const float* __restrict__ cost,
    bf16_t* __restrict__ ctxb)
{
    __shared__ float sc[16 * 1024];            // exactly 64 KB

    const int lane = threadIdx.x;
    const int tile = blockIdx.x;               // 0..4095
    const int q16  = tile & 63;
    const int h    = (tile >> 6) & 7;
    const int b    = tile >> 9;
    const int q0   = q16 * 16;
    const size_t bq = (size_t)b * 1024;

    const bf16_t* Q  = qb + (bq + q0) * 512 + h * 64;   // lda = 512
    const bf16_t* Kt = kb + bq * 512 + h * 64;          // row n at n*512

    const int nlane = lane & 15;
    const int rb    = (lane >> 4) << 3;

    // ---- scores: QK^T (+cost) into LDS -----------------------------------
    v16bf aq0 = load_a(Q, 512, 0, 0,  lane);
    v16bf aq1 = load_a(Q, 512, 0, 32, lane);
    for (int nt = 0; nt < 64; ++nt) {
        int n0 = nt * 16;
        v8f acc = {};
        acc = wmma_bf16(aq0, load_b(Kt, 512, n0, 0,  lane), acc);
        acc = wmma_bf16(aq1, load_b(Kt, 512, n0, 32, lane), acc);
        int n = n0 + nlane;
        const float* cp = cost + (bq + q0 + rb) * 1024 + n;
#pragma unroll
        for (int r = 0; r < 8; ++r)
            sc[(rb + r) * 1024 + n] = acc[r] + cp[(size_t)r * 1024];
    }
    __syncthreads();

    // ---- softmax: lane l owns row l&15, half (l>>4) ----------------------
    const int row = lane & 15;
    const int hh  = lane >> 4;
    float mx = -3.0e38f;
    {
        const float* p = sc + row * 1024 + hh * 512;
        for (int i = 0; i < 512; ++i) mx = fmaxf(mx, p[i]);
    }
    mx = fmaxf(mx, __shfl_xor(mx, 16, 32));
    __syncthreads();

    float s = 0.0f;
    {
        volatile float*    p  = sc + row * 1024 + hh * 512;
        volatile uint16_t* pw = (volatile uint16_t*)(sc + row * 1024) + hh * 1024;
        for (int i = 0; i < 512; ++i) {
            float x = p[i];                    // read byte 4i of this region
            float e = __expf(x - mx);
            s += e;
            pw[i] = __builtin_bit_cast(uint16_t, f2bf(e));  // write byte 2i
        }
    }
    float stot = s + __shfl_xor(s, 16, 32);
    // stash row sums in dead LDS bytes (last f32 slot of each row is unused
    // by the bf16 prob layout)
    if (lane < 16) sc[lane * 1024 + 1023] = stot;
    __syncthreads();

    float invs[8];
#pragma unroll
    for (int r = 0; r < 8; ++r) invs[r] = 1.0f / sc[(rb + r) * 1024 + 1023];

    // ---- ctx = P V : A from LDS probs, B from Vt[b,h,d,s] ----------------
    const bf16_t* V = vt + ((size_t)(b * 8 + h) * 64) * 1024;  // ldb = 1024
    for (int dt = 0; dt < 4; ++dt) {
        v8f acc = {};
        for (int k0 = 0; k0 < 1024; k0 += 32) {
            v16bf a  = load_a_lds((const bf16_t*)sc, k0, lane);
            v16bf bb = load_b(V, 1024, dt * 16, k0, lane);
            acc = wmma_bf16(a, bb, acc);
        }
        int d = dt * 16 + nlane;
        bf16_t* op = ctxb + (bq + q0 + rb) * 512 + h * 64 + d;
#pragma unroll
        for (int r = 0; r < 8; ++r)
            op[(size_t)r * 512] = f2bf(acc[r] * invs[r]);
    }
}

// ---------------------------------------------------------------------------
// LayerNorm over D=512, one wave per row; optional f32 + bf16 outputs.
// ---------------------------------------------------------------------------
__global__ __launch_bounds__(128) void ln_kernel(
    const float* __restrict__ in, const float* __restrict__ g,
    const float* __restrict__ bb, float* __restrict__ outf,
    bf16_t* __restrict__ outb, int rows)
{
    int lane = threadIdx.x & 31;
    int wave = threadIdx.x >> 5;
    int row  = blockIdx.x * 4 + wave;
    if (row >= rows) return;
    const float* p = in + (size_t)row * 512;
    float xr[16];
    float s = 0.0f, sq = 0.0f;
#pragma unroll
    for (int i = 0; i < 16; ++i) {
        float x = p[lane + 32 * i];
        xr[i] = x; s += x; sq += x * x;
    }
#pragma unroll
    for (int off = 16; off > 0; off >>= 1) {
        s  += __shfl_xor(s,  off, 32);
        sq += __shfl_xor(sq, off, 32);
    }
    float mu  = s  * (1.0f / 512.0f);
    float var = sq * (1.0f / 512.0f) - mu * mu;
    float k   = rsqrtf(var + 1e-6f);
#pragma unroll
    for (int i = 0; i < 16; ++i) {
        int d = lane + 32 * i;
        float y = (xr[i] - mu) * k * g[d] + bb[d];
        if (outf) outf[(size_t)row * 512 + d] = y;
        if (outb) outb[(size_t)row * 512 + d] = f2bf(y);
    }
}

// ---------------------------------------------------------------------------
// Orchestration
// ---------------------------------------------------------------------------
extern "C" void kernel_launch(void* const* d_in, const int* in_sizes, int n_in,
                              void* d_out, int out_size, void* d_ws, size_t ws_size,
                              hipStream_t stream) {
    const float* enc  = (const float*)d_in[0];
    const float* cost = (const float*)d_in[1];
    const float* wq   = (const float*)d_in[2];
    const float* wk   = (const float*)d_in[3];
    const float* wv   = (const float*)d_in[4];
    const float* fcw  = (const float*)d_in[5];
    const float* ln1g = (const float*)d_in[6];
    const float* ln1b = (const float*)d_in[7];
    const float* w1   = (const float*)d_in[8];
    const float* b1   = (const float*)d_in[9];
    const float* w2   = (const float*)d_in[10];
    const float* b2   = (const float*)d_in[11];
    const float* ln2g = (const float*)d_in[12];
    const float* ln2b = (const float*)d_in[13];
    float* out = (float*)d_out;

    char* ws = (char*)d_ws;
    const size_t MB = 1ull << 20;
    bf16_t* xb   = (bf16_t*)(ws + 0);                 //  8 MB enc bf16
    bf16_t* qb   = (bf16_t*)(ws +  8 * MB);           //  8 MB
    bf16_t* kb   = (bf16_t*)(ws + 16 * MB);           //  8 MB
    bf16_t* vt   = (bf16_t*)(ws + 24 * MB);           //  8 MB V^T
    bf16_t* vbuf = (bf16_t*)(ws + 32 * MB);           //  8 MB V, then ctx
    float*  tmp  = (float*) (ws + 40 * MB);           // 16 MB pre-LN f32
    float*  l1f  = (float*) (ws + 56 * MB);           // 16 MB attn_out f32
    bf16_t* l1b  = (bf16_t*)(ws + 72 * MB);           //  8 MB attn_out bf16
    bf16_t* wsm  = (bf16_t*)(ws + 80 * MB);           // 512 KB small weight
    bf16_t* wbig = (bf16_t*)(ws + 80 * MB + 512 * 1024); // 2 MB big weight
    bf16_t* hbuf = (bf16_t*)(ws +  8 * MB);           // 32 MB, overlays qb..vbuf

    const int Mrows = 8192, D = 512, DF = 2048;
    const int nEl  = Mrows * D;      // 4,194,304
    const int wElS = D * D;          // 262,144
    const int wElB = DF * D;         // 1,048,576
    dim3 blk(256);
    dim3 gN512(512 / 64,  Mrows / 128);   // (8, 64)
    dim3 gN2048(2048 / 64, Mrows / 128);  // (32, 64)

    // enc -> bf16
    cast_f32_bf16<<<(nEl + 255) / 256, blk, 0, stream>>>(enc, xb, nEl);

    // Q = enc wq^T (scaled by 1/sqrt(64)), K, V
    cast_f32_bf16<<<(wElS + 255) / 256, blk, 0, stream>>>(wq, wsm, wElS);
    gemm_bf16_wmma<<<gN512, blk, 0, stream>>>(xb, wsm, nullptr, nullptr,
                                              nullptr, qb, Mrows, D, D, 0, 0.125f);
    cast_f32_bf16<<<(wElS + 255) / 256, blk, 0, stream>>>(wk, wsm, wElS);
    gemm_bf16_wmma<<<gN512, blk, 0, stream>>>(xb, wsm, nullptr, nullptr,
                                              nullptr, kb, Mrows, D, D, 0, 1.0f);
    cast_f32_bf16<<<(wElS + 255) / 256, blk, 0, stream>>>(wv, wsm, wElS);
    gemm_bf16_wmma<<<gN512, blk, 0, stream>>>(xb, wsm, nullptr, nullptr,
                                              nullptr, vbuf, Mrows, D, D, 0, 1.0f);
    transpose_v<<<(nEl + 255) / 256, blk, 0, stream>>>(vbuf, vt);

    // fused attention -> ctx bf16 (reuses vbuf)
    attn_fused<<<4096, dim3(32), 0, stream>>>(qb, kb, vt, cost, vbuf);

    // fc projection + residual(enc) -> tmp ; LN1 -> l1f/l1b
    cast_f32_bf16<<<(wElS + 255) / 256, blk, 0, stream>>>(fcw, wsm, wElS);
    gemm_bf16_wmma<<<gN512, blk, 0, stream>>>(vbuf, wsm, nullptr, enc,
                                              tmp, nullptr, Mrows, D, D, 0, 1.0f);
    ln_kernel<<<Mrows / 4, dim3(128), 0, stream>>>(tmp, ln1g, ln1b, l1f, l1b, Mrows);

    // FFN1: relu(l1 w1^T + b1) -> hbuf bf16
    cast_f32_bf16<<<(wElB + 255) / 256, blk, 0, stream>>>(w1, wbig, wElB);
    gemm_bf16_wmma<<<gN2048, blk, 0, stream>>>(l1b, wbig, b1, nullptr,
                                               nullptr, hbuf, Mrows, DF, D, 1, 1.0f);

    // FFN2: h w2^T + b2 + residual(l1f) -> tmp ; LN2 -> out
    cast_f32_bf16<<<(wElB + 255) / 256, blk, 0, stream>>>(w2, wbig, wElB);
    gemm_bf16_wmma<<<gN512, blk, 0, stream>>>(hbuf, wbig, b2, l1f,
                                              tmp, nullptr, Mrows, D, DF, 0, 1.0f);
    ln_kernel<<<Mrows / 4, dim3(128), 0, stream>>>(tmp, ln2g, ln2b, out, nullptr, Mrows);
}
